// DemoGraphNet_32830730011147
// MI455X (gfx1250) — compile-verified, matrix operand order checked by reading, not run on hardware
//
#include <hip/hip_runtime.h>

#define N_NODES   100000
#define N_EDGES   1600000
#define NUM_GRAPHS 256
#define HID       128
#define OUT_C     8

typedef _Float16 v16h __attribute__((ext_vector_type(16)));
typedef _Float16 v4h  __attribute__((ext_vector_type(4)));
typedef float    v8f  __attribute__((ext_vector_type(8)));
typedef float    v4f  __attribute__((ext_vector_type(4)));

union FragH { v16h v; v4f f4[2]; };

// ---------------- degree / normalization ----------------
__global__ __launch_bounds__(256) void k_init_deg(float* deg) {
    int i = blockIdx.x * 256 + threadIdx.x;
    if (i < N_NODES) deg[i] = 1.0f;                    // self-loop
}

__global__ __launch_bounds__(256) void k_edge_deg(const int* __restrict__ ei, float* deg) {
    int e = blockIdx.x * 256 + threadIdx.x;
    if (e < N_EDGES) unsafeAtomicAdd(&deg[ei[N_EDGES + e]], 1.0f);
}

__global__ __launch_bounds__(256) void k_rsqrt(float* deg) {
    int i = blockIdx.x * 256 + threadIdx.x;
    if (i < N_NODES) deg[i] = rsqrtf(deg[i]);          // in-place -> inv_sqrt
}

// ---------------- f32 -> f16 conversions ----------------
__global__ __launch_bounds__(256) void k_f32_to_f16(const float* __restrict__ in,
                                                    _Float16* __restrict__ out, int n4) {
    int i = blockIdx.x * 256 + threadIdx.x;
    if (i < n4) {
        v4f v = reinterpret_cast<const v4f*>(in)[i];
        reinterpret_cast<v4h*>(out)[i] = __builtin_convertvector(v, v4h);
    }
}

// Wt[n][k] = W[k][n] as f16 (column-major weight for the B fragment)
__global__ __launch_bounds__(256) void k_transpose_w(const float* __restrict__ W,
                                                     _Float16* __restrict__ Wt) {
    int i = blockIdx.x * 256 + threadIdx.x;            // i = n*128 + k, 16384 total
    int n = i >> 7, k = i & 127;
    Wt[i] = (_Float16)W[k * HID + n];
}

// ---------------- WMMA GEMM: H[M,128] = A[M,128] @ W[128,128] ----------------
// block = 256 thr = 8 waves; block -> 16 rows, wave w -> columns [16w,16w+16)
__global__ __launch_bounds__(256) void k_gemm_f16(const _Float16* __restrict__ A,
                                                  const _Float16* __restrict__ Wt,
                                                  float* __restrict__ H) {
    const int rowBase = blockIdx.x * 16;
    const int lane    = threadIdx.x & 31;
    const int n0      = (threadIdx.x >> 5) * 16;
    const int r       = lane & 15;
    const int kb      = (lane >> 4) * 8;     // K sub-offset within a 32-wide step

    const v4f* ap = reinterpret_cast<const v4f*>(A  + (rowBase + r) * HID);
    const v4f* bp = reinterpret_cast<const v4f*>(Wt + (n0      + r) * HID);

    FragH a[4], b[4];
#pragma unroll
    for (int kt = 0; kt < 4; ++kt) {
        int ci = (kt * 32 + kb) >> 3;        // float4 index within the 128-half row
        a[kt].f4[0] = ap[ci];  a[kt].f4[1] = ap[ci + 2];   // K: kb..kb+7, kb+16..kb+23
        b[kt].f4[0] = bp[ci];  b[kt].f4[1] = bp[ci + 2];
    }

    v8f c = {};
#pragma unroll
    for (int kt = 0; kt < 4; ++kt)
        c = __builtin_amdgcn_wmma_f32_16x16x32_f16(false, a[kt].v, false, b[kt].v,
                                                   (short)0, c, false, false);

    const int m0 = rowBase + (lane >> 4) * 8;
    const int n  = n0 + r;
#pragma unroll
    for (int rr = 0; rr < 8; ++rr)
        H[(m0 + rr) * HID + n] = c[rr];      // raw A@W, bias added post-aggregation
}

// ---------------- aggregation ----------------
// agg[i] = H[i] * inv_sqrt[i]^2   (self-loop term), 32 threads per node, float4 lanes
__global__ __launch_bounds__(256) void k_self_init(const float* __restrict__ H,
                                                   const float* __restrict__ invs,
                                                   float* __restrict__ agg) {
    int t = blockIdx.x * 256 + threadIdx.x;
    int i = t >> 5, cg = (t & 31) * 4;
    float s = invs[i]; s *= s;
    v4f h = *reinterpret_cast<const v4f*>(H + i * HID + cg);
    *reinterpret_cast<v4f*>(agg + i * HID + cg) = h * s;
}

// agg[dst] += H[src] * invs[src]*invs[dst], one wave (32 lanes x float4) per edge
__global__ __launch_bounds__(256) void k_edge_scatter(const int* __restrict__ ei,
                                                      const float* __restrict__ invs,
                                                      const float* __restrict__ H,
                                                      float* __restrict__ agg) {
    int t = blockIdx.x * 256 + threadIdx.x;
    int e = t >> 5, cg = (t & 31) * 4;
    int s = ei[e], d = ei[N_EDGES + e];
    float w = invs[s] * invs[d];
    v4f h = *reinterpret_cast<const v4f*>(H + s * HID + cg);
    float* out = agg + d * HID + cg;
    unsafeAtomicAdd(out + 0, h.x * w);
    unsafeAtomicAdd(out + 1, h.y * w);
    unsafeAtomicAdd(out + 2, h.z * w);
    unsafeAtomicAdd(out + 3, h.w * w);
}

// relu(agg + b) -> optional f32 copy and/or f16 copy (next layer's GEMM input)
__global__ __launch_bounds__(256) void k_finish(const float* __restrict__ agg,
                                                const float* __restrict__ bias,
                                                float* __restrict__ f32out,
                                                _Float16* __restrict__ f16out) {
    int t = blockIdx.x * 256 + threadIdx.x;
    int i = t >> 5, cg = (t & 31) * 4;
    v4f v = *reinterpret_cast<const v4f*>(agg + i * HID + cg);
    v4f bb = *reinterpret_cast<const v4f*>(bias + cg);
    v = v + bb;
    v.x = fmaxf(v.x, 0.0f); v.y = fmaxf(v.y, 0.0f);
    v.z = fmaxf(v.z, 0.0f); v.w = fmaxf(v.w, 0.0f);
    if (f32out) *reinterpret_cast<v4f*>(f32out + i * HID + cg) = v;
    if (f16out) *reinterpret_cast<v4h*>(f16out + i * HID + cg) =
                    __builtin_convertvector(v, v4h);
}

// ---------------- pooling + head ----------------
__global__ __launch_bounds__(256) void k_pool(const int* __restrict__ batch,
                                              const float* __restrict__ h,
                                              float* __restrict__ pooled,
                                              float* __restrict__ cnt) {
    int t = blockIdx.x * 256 + threadIdx.x;
    int i = t >> 5, cg = (t & 31) * 4;
    int g = batch[i];
    v4f v = *reinterpret_cast<const v4f*>(h + i * HID + cg);
    float* p = pooled + g * HID + cg;
    unsafeAtomicAdd(p + 0, v.x);
    unsafeAtomicAdd(p + 1, v.y);
    unsafeAtomicAdd(p + 2, v.z);
    unsafeAtomicAdd(p + 3, v.w);
    if ((t & 31) == 0) unsafeAtomicAdd(cnt + g, 1.0f);
}

__global__ __launch_bounds__(128) void k_head(const float* __restrict__ pooled,
                                              const float* __restrict__ cnt,
                                              const float* __restrict__ Wh1,
                                              const float* __restrict__ bh1,
                                              const float* __restrict__ Wh2,
                                              const float* __restrict__ bh2,
                                              float* __restrict__ out) {
    __shared__ float mean[HID];
    __shared__ float hid[HID];
    int g = blockIdx.x, j = threadIdx.x;
    float c = cnt[g]; c = c > 1.0f ? c : 1.0f;
    mean[j] = pooled[g * HID + j] / c;
    __syncthreads();
    float acc = bh1[j];
    for (int k = 0; k < HID; ++k) acc += mean[k] * Wh1[k * HID + j];
    hid[j] = fmaxf(acc, 0.0f);
    __syncthreads();
    if (j < OUT_C) {
        float o = bh2[j];
        for (int k = 0; k < HID; ++k) o += hid[k] * Wh2[k * OUT_C + j];
        out[g * OUT_C + j] = o;
    }
}

extern "C" void kernel_launch(void* const* d_in, const int* in_sizes, int n_in,
                              void* d_out, int out_size, void* d_ws, size_t ws_size,
                              hipStream_t stream) {
    const float* x     = (const float*)d_in[0];
    const int*   ei    = (const int*)d_in[1];
    const int*   batch = (const int*)d_in[2];
    const float* W1    = (const float*)d_in[3];
    const float* b1    = (const float*)d_in[4];
    const float* W2    = (const float*)d_in[5];
    const float* b2    = (const float*)d_in[6];
    const float* Wh1   = (const float*)d_in[7];
    const float* bh1   = (const float*)d_in[8];
    const float* Wh2   = (const float*)d_in[9];
    const float* bh2   = (const float*)d_in[10];
    float* out = (float*)d_out;

    // workspace carving (256B aligned)
    char* base = (char*)d_ws;
    size_t off = 0;
    auto carve = [&](size_t bytes) -> char* {
        char* p = base + off;
        off = (off + bytes + 255) & ~(size_t)255;
        return p;
    };
    float*    invs   = (float*)   carve((size_t)N_NODES * 4);            // deg -> inv_sqrt
    _Float16* xh     = (_Float16*)carve((size_t)N_NODES * HID * 2);      // f16 activations
    _Float16* w1t    = (_Float16*)carve((size_t)HID * HID * 2);
    _Float16* w2t    = (_Float16*)carve((size_t)HID * HID * 2);
    float*    H      = (float*)   carve((size_t)N_NODES * HID * 4);      // A@W / final h2
    float*    agg    = (float*)   carve((size_t)N_NODES * HID * 4);      // accumulator
    float*    pooled = (float*)   carve((size_t)NUM_GRAPHS * HID * 4 + NUM_GRAPHS * 4);
    float*    cnt    = pooled + NUM_GRAPHS * HID;

    const int nodeB  = (N_NODES + 255) / 256;            // 391
    const int edgeB  = N_EDGES / 256;                    // 6250
    const int featB  = (N_NODES * HID / 4) / 256;        // 12500 (vec4 elementwise)
    const int nodeWB = (N_NODES * 32) / 256;             // 12500 (32 thr/node)
    const int edgeWB = (N_EDGES * 32) / 256;             // 200000 (32 thr/edge)
    const int gemmB  = N_NODES / 16;                     // 6250

    // normalization coefficients
    k_init_deg<<<nodeB, 256, 0, stream>>>(invs);
    k_edge_deg<<<edgeB, 256, 0, stream>>>(ei, invs);
    k_rsqrt  <<<nodeB, 256, 0, stream>>>(invs);

    // f16 operands
    k_f32_to_f16 <<<featB, 256, 0, stream>>>(x, xh, N_NODES * HID / 4);
    k_transpose_w<<<64,    256, 0, stream>>>(W1, w1t);
    k_transpose_w<<<64,    256, 0, stream>>>(W2, w2t);

    // ---- layer 1 ----
    k_gemm_f16    <<<gemmB,  256, 0, stream>>>(xh, w1t, H);
    k_self_init   <<<nodeWB, 256, 0, stream>>>(H, invs, agg);
    k_edge_scatter<<<edgeWB, 256, 0, stream>>>(ei, invs, H, agg);
    k_finish      <<<nodeWB, 256, 0, stream>>>(agg, b1, (float*)nullptr, xh);

    // ---- layer 2 ----
    k_gemm_f16    <<<gemmB,  256, 0, stream>>>(xh, w2t, H);
    k_self_init   <<<nodeWB, 256, 0, stream>>>(H, invs, agg);
    k_edge_scatter<<<edgeWB, 256, 0, stream>>>(ei, invs, H, agg);
    k_finish      <<<nodeWB, 256, 0, stream>>>(agg, b2, H, (_Float16*)nullptr);

    // ---- pool + head ----
    hipMemsetAsync(pooled, 0, (size_t)(NUM_GRAPHS * HID + NUM_GRAPHS) * 4, stream);
    k_pool<<<nodeWB, 256, 0, stream>>>(batch, H, pooled, cnt);
    k_head<<<NUM_GRAPHS, 128, 0, stream>>>(pooled, cnt, Wh1, bh1, Wh2, bh2, out);
}
